// L2CrossAttention_88940182765582
// MI455X (gfx1250) — compile-verified, matrix operand order checked by reading, not run on hardware
//
#include <hip/hip_runtime.h>

// ---------------------------------------------------------------------------
// L2 self-attention for MI455X (gfx1250, wave32, WMMA bf16 16x16x32)
//   q = x@Wq+bq ; v = x@Wv+bv          (bf16 GEMM, f32 accum)
//   sim = -(AA - 2AB + BB)/8  (<= 0, max == 0  ->  no softmax rescale needed)
//   O = exp(sim) @ V ; L = exp(sim) @ 1 ; attn_out = O / L   (all via WMMA)
//   out = attn_out @ Wo + bo
// Uses CDNA5 async global->LDS DMA (ASYNCcnt) for pure-copy tile staging.
// ---------------------------------------------------------------------------

typedef __bf16 bf16;
typedef __attribute__((ext_vector_type(16))) __bf16 v16bf;
typedef __attribute__((ext_vector_type(8)))  float  v8f;
typedef __attribute__((ext_vector_type(4)))  unsigned int v4u;

union Frag {          // one 16x32 bf16 WMMA operand: 8 VGPRs = 2 x 16B chunks
  v16bf bf;
  v4u   q[2];
};

constexpr int Bsz = 4, Nseq = 2048, Dmod = 1024, NH = 16, Dh = 64, Inner = 1024;
constexpr float kScale = 0.125f;                  // 1/sqrt(64)
constexpr float kLog2e = 1.4426950408889634f;

static __device__ __forceinline__ v8f wmma_bf16(const Frag& a, const Frag& b, v8f c) {
  return __builtin_amdgcn_wmma_f32_16x16x32_bf16(false, a.bf, false, b.bf,
                                                 (short)0, c, false, false);
}

// Async 16-byte global->LDS copy (CDNA5 TDM-lite path, tracked by ASYNCcnt).
// Low 32 bits of a generic __shared__ pointer are the LDS byte address.
static __device__ __forceinline__ void async_cp16(const bf16* gsrc, bf16* ldst) {
  const unsigned lds  = (unsigned)(size_t)ldst;
  const unsigned long long gp = (unsigned long long)(size_t)gsrc;
  asm volatile("global_load_async_to_lds_b128 %0, %1, off"
               :: "v"(lds), "v"(gp) : "memory");
}
static __device__ __forceinline__ void async_wait0() {
  asm volatile("s_wait_asynccnt 0x0" ::: "memory");
}

// ---------------------------------------------------------------------------
// Tiled GEMM with bias:  Out[M x Nout] = A[M x K] @ W[K x Nout] + bias
// 128x128 tile / workgroup, K-tile 64, 8 waves, wave = 16x128 strip.
// A fp32 (convert) or bf16 (async-copied); W/bias fp32; Out bf16 or fp32.
// ---------------------------------------------------------------------------
template <bool A_BF16, bool OUT_BF16>
__global__ __launch_bounds__(256)
void gemm_bias_kernel(const void* __restrict__ Ap, const float* __restrict__ W,
                      const float* __restrict__ bias, void* __restrict__ Outp,
                      int K, int Nout) {
  __shared__ __align__(16) bf16 As[128][72];   // 128x64 tile, +8 pad (16B mult)
  __shared__ __align__(16) bf16 Bs[128][72];   // W^T tile: Bs[n][k]

  const int tid  = threadIdx.x;
  const int wave = tid >> 5, lane = tid & 31, l = lane & 15, grp = lane >> 4;
  const int n0   = blockIdx.x * 128;
  const long i0  = (long)blockIdx.y * 128;

  v8f acc[8];
#pragma unroll
  for (int t = 0; t < 8; ++t) acc[t] = (v8f)0.0f;

  for (int k0 = 0; k0 < K; k0 += 64) {
    __syncthreads();
    if constexpr (A_BF16) {                    // pure copy -> async DMA to LDS
#pragma unroll
      for (int t = 0; t < 4; ++t) {            // 128 rows x 8 chunks of 16B
        int idx = tid + t * 256;
        int r = idx >> 3, c8 = (idx & 7) * 8;
        async_cp16(((const bf16*)Ap) + (i0 + r) * K + k0 + c8, &As[r][c8]);
      }
    } else {                                   // fp32 -> bf16 conversion copy
#pragma unroll
      for (int t = 0; t < 32; ++t) {
        int idx = tid + t * 256;
        int r = idx >> 6, c = idx & 63;
        As[r][c] = (bf16)((const float*)Ap)[(i0 + r) * K + k0 + c];
      }
    }
#pragma unroll
    for (int t = 0; t < 32; ++t) {             // stage W tile transposed
      int idx = tid + t * 256;
      int k = idx >> 7, n = idx & 127;
      Bs[n][k] = (bf16)W[(long)(k0 + k) * Nout + n0 + n];
    }
    if constexpr (A_BF16) async_wait0();
    __syncthreads();

#pragma unroll
    for (int kc = 0; kc < 2; ++kc) {
      Frag a;                                  // A fragment (rows wave*16..+15)
      const bf16* ar = &As[wave * 16 + l][kc * 32 + grp * 8];
      a.q[0] = *(const v4u*)ar;                // K = grp*8 .. +7
      a.q[1] = *(const v4u*)(ar + 16);         // K = grp*8+16 .. +23
#pragma unroll
      for (int nt = 0; nt < 8; ++nt) {
        Frag b;                                // B fragment: per-lane column
        const bf16* br = &Bs[nt * 16 + l][kc * 32 + grp * 8];
        b.q[0] = *(const v4u*)br;
        b.q[1] = *(const v4u*)(br + 16);
        acc[nt] = wmma_bf16(a, b, acc[nt]);
      }
    }
  }

#pragma unroll
  for (int nt = 0; nt < 8; ++nt) {
    const int col = n0 + nt * 16 + l;
    const float bv = bias[col];
#pragma unroll
    for (int r = 0; r < 8; ++r) {              // C layout: VGPR r -> M = r + grp*8
      const long row = i0 + wave * 16 + r + grp * 8;
      const float o = acc[nt][r] + bv;
      if constexpr (OUT_BF16)
        ((bf16*)Outp)[row * Nout + col] = (bf16)o;
      else
        ((float*)Outp)[row * Nout + col] = o;
    }
  }
}

// ---------------------------------------------------------------------------
// bb[b,h,i] = sum_d q[b,i,h*64+d]^2   (the AA / BB vector; k == q here)
// ---------------------------------------------------------------------------
__global__ __launch_bounds__(256)
void rowsq_kernel(const bf16* __restrict__ qb, float* __restrict__ bb) {
  const int i = blockIdx.x * 256 + threadIdx.x;   // flat (b, n, h)
  const int h = i & (NH - 1);
  const int n = (i >> 4) & (Nseq - 1);
  const int b = i >> 15;
  const bf16* p = qb + ((long)(b * Nseq + n)) * Inner + h * Dh;
  float s = 0.f;
#pragma unroll
  for (int d = 0; d < Dh; ++d) { const float q = (float)p[d]; s += q * q; }
  bb[((long)(b * NH + h)) * Nseq + n] = s;
}

// ---------------------------------------------------------------------------
// Attention: workgroup per (b*h, 128-row strip); wave w owns 16 rows.
// Per 128-col block: async-stage Q_j (pure copy) + V^T + BB into LDS, then
// 16 WMMA for S = Q Q^T, exp -> P (per-wave LDS transpose), 4 WMMA for L
// (ones trick), 16 WMMA for O += P V.  L tile is elementwise-aligned with O.
// ---------------------------------------------------------------------------
__global__ __launch_bounds__(256)
void attn_kernel(const bf16* __restrict__ qb, const bf16* __restrict__ vb,
                 const float* __restrict__ bbuf, bf16* __restrict__ ob) {
  __shared__ __align__(16) bf16  qs[128][72];      // Q_j tile row-major, 18 KB
  __shared__ __align__(16) bf16  vt[Dh][136];      // V^T tile vt[d][j], 17 KB
  __shared__ __align__(16) bf16  ps[8][16][136];   // per-wave P scratch, 34 KB
  __shared__ __align__(16) float bbs[128];         // BB block

  const int tid  = threadIdx.x;
  const int wave = tid >> 5, lane = tid & 31, l = lane & 15, grp = lane >> 4;
  const int b = blockIdx.x >> 4, h = blockIdx.x & 15;
  const int iW = blockIdx.y * 128 + wave * 16;
  const bf16* qbBH = qb + ((long)b * Nseq) * Inner + h * Dh;   // row i stride Inner
  const bf16* vbBH = vb + ((long)b * Nseq) * Inner + h * Dh;

  // A fragments of this wave's 16 Q rows (K = d, two 32-chunks), loaded once.
  Frag aQ[2];
  {
    const bf16* qr = qbBH + (long)(iW + l) * Inner;
#pragma unroll
    for (int c = 0; c < 2; ++c) {
      const bf16* pch = qr + c * 32 + grp * 8;
      aQ[c].q[0] = *(const v4u*)pch;
      aQ[c].q[1] = *(const v4u*)(pch + 16);
    }
  }
  const float* bbBH = bbuf + ((long)(b * NH + h)) * Nseq;
  float aa[8];
#pragma unroll
  for (int r = 0; r < 8; ++r) aa[r] = bbBH[iW + r + grp * 8];

  Frag ones;                                       // all-halves bf16 1.0
  ones.q[0] = (v4u)0x3F803F80u;
  ones.q[1] = (v4u)0x3F803F80u;

  v8f oacc[4];
#pragma unroll
  for (int t = 0; t < 4; ++t) oacc[t] = (v8f)0.0f;
  v8f lacc = (v8f)0.0f;

  for (int j0 = 0; j0 < Nseq; j0 += 128) {
    __syncthreads();                               // LDS reuse barrier
#pragma unroll
    for (int t = 0; t < 4; ++t) {                  // Q_j tile: async DMA, 16 KB
      int idx = tid + t * 256;
      int r = idx >> 3, c8 = (idx & 7) * 8;
      async_cp16(qbBH + (long)(j0 + r) * Inner + c8, &qs[r][c8]);
    }
#pragma unroll
    for (int t = 0; t < 32; ++t) {                 // V^T tile (transposing copy)
      int idx = tid + t * 256;
      int j = idx >> 6, d = idx & 63;
      vt[d][j] = vbBH[(long)(j0 + j) * Inner + d];
    }
    if (tid < 128) bbs[tid] = bbBH[j0 + tid];      // BB block
    if (j0 + 128 < Nseq) {                         // global_prefetch_b8 next block
      int j = tid >> 1, d = (tid & 1) * 32;
      __builtin_prefetch(vbBH + (long)(j0 + 128 + j) * Inner + d, 0, 1);
    }
    async_wait0();
    __syncthreads();

#pragma unroll
    for (int jt = 0; jt < 8; ++jt) {
      // B fragment of Q_j from LDS: same per-lane pattern as A frags (Q Q^T).
      const bf16* qjr = &qs[jt * 16 + l][grp * 8];
      v8f s = (v8f)0.0f;
#pragma unroll
      for (int c = 0; c < 2; ++c) {
        Frag bq;
        bq.q[0] = *(const v4u*)(qjr + c * 32);
        bq.q[1] = *(const v4u*)(qjr + c * 32 + 16);
        s = wmma_bf16(aQ[c], bq, s);               // S += Q_i (Q_j)^T
      }
      const float bbv = bbs[jt * 16 + l];          // column constant BB_j
#pragma unroll
      for (int r = 0; r < 8; ++r) {                // P = exp2(log2e*scale*sim)
        const float e = (2.f * s[r] - aa[r] - bbv) * (kScale * kLog2e);
        ps[wave][r + grp * 8][jt * 16 + l] = (bf16)__builtin_amdgcn_exp2f(e);
      }
    }

    // O += P @ V  and  L += P @ ones   (A frags of P from per-wave LDS)
#pragma unroll
    for (int kc = 0; kc < 4; ++kc) {
      Frag aP;
      const bf16* pr = &ps[wave][l][kc * 32 + grp * 8];
      aP.q[0] = *(const v4u*)pr;
      aP.q[1] = *(const v4u*)(pr + 16);
      lacc = wmma_bf16(aP, ones, lacc);
#pragma unroll
      for (int dt = 0; dt < 4; ++dt) {
        Frag bV;
        const bf16* vr = &vt[dt * 16 + l][kc * 32 + grp * 8];
        bV.q[0] = *(const v4u*)vr;
        bV.q[1] = *(const v4u*)(vr + 16);
        oacc[dt] = wmma_bf16(aP, bV, oacc[dt]);
      }
    }
  }

#pragma unroll
  for (int dt = 0; dt < 4; ++dt) {
#pragma unroll
    for (int r = 0; r < 8; ++r) {                  // normalize + store bf16
      const long row = (long)b * Nseq + iW + r + grp * 8;
      ob[row * Inner + h * Dh + dt * 16 + l] = (bf16)(oacc[dt][r] / lacc[r]);
    }
  }
}

// ---------------------------------------------------------------------------
extern "C" void kernel_launch(void* const* d_in, const int* in_sizes, int n_in,
                              void* d_out, int out_size, void* d_ws, size_t ws_size,
                              hipStream_t stream) {
  (void)in_sizes; (void)n_in; (void)out_size; (void)ws_size;
  const float* x  = (const float*)d_in[0];
  const float* Wq = (const float*)d_in[1];
  const float* bq = (const float*)d_in[2];
  const float* Wv = (const float*)d_in[3];
  const float* bv = (const float*)d_in[4];
  const float* Wo = (const float*)d_in[5];
  const float* bo = (const float*)d_in[6];

  const size_t tokens = (size_t)Bsz * Nseq;        // 8192
  bf16*  qbuf = (bf16*)d_ws;                       // 16 MB
  bf16*  vbuf = qbuf + tokens * Inner;             // 16 MB
  bf16*  obuf = vbuf + tokens * Inner;             // 16 MB
  float* bbuf = (float*)(obuf + tokens * Inner);   // 512 KB

  dim3 blk(256);
  dim3 gProj(Inner / 128, tokens / 128);           // 8 x 64
  gemm_bias_kernel<false, true><<<gProj, blk, 0, stream>>>(x, Wq, bq, qbuf, Dmod, Inner);
  gemm_bias_kernel<false, true><<<gProj, blk, 0, stream>>>(x, Wv, bv, vbuf, Dmod, Inner);
  rowsq_kernel<<<dim3((Bsz * Nseq * NH) / 256), blk, 0, stream>>>(qbuf, bbuf);
  attn_kernel<<<dim3(Bsz * NH, Nseq / 128), blk, 0, stream>>>(qbuf, vbuf, bbuf, obuf);
  gemm_bias_kernel<true, false><<<dim3(Dmod / 128, tokens / 128), blk, 0, stream>>>(
      obuf, Wo, bo, (float*)d_out, Inner, Dmod);
}
